// WindowedSelfAttentionLayer_9801115370181
// MI455X (gfx1250) — matrix operand order repaired and checked by reading, not compiled, not run on hardware
//
#include <hip/hip_runtime.h>

#define Bsz   2
#define Ssz   2048
#define Dsz   1024
#define Hsz   16
#define DHsz  64
#define HIDsz 4096
#define HALFW 128

typedef __attribute__((ext_vector_type(16))) _Float16 v16h;
typedef __attribute__((ext_vector_type(8)))  _Float16 v8h;
typedef __attribute__((ext_vector_type(8)))  float    v8f;
typedef __attribute__((ext_vector_type(4)))  int      v4i;

union AFrag { v16h v; v8h h[2]; };

// CDNA5 async global->LDS copy (ASYNCcnt-tracked), when the toolchain has it.
#if defined(__has_builtin)
#if __has_builtin(__builtin_amdgcn_global_load_async_to_lds_b128)
#define ASYNC_LDS 1
#endif
#endif
#ifndef ASYNC_LDS
#define ASYNC_LDS 0
#endif

#if ASYNC_LDS
// Diagnostic from this toolchain shows param0 is a non-const AS1 (global,
// printed "__device__") pointer to int4. Param1 is the LDS-side pointer.
typedef v4i __attribute__((address_space(1)))* gv4i_p;
typedef v4i __attribute__((address_space(3)))* lv4i_p;
#endif

__device__ __forceinline__ void wait_async0() {
#if defined(__has_builtin)
#if __has_builtin(__builtin_amdgcn_s_wait_asynccnt)
  __builtin_amdgcn_s_wait_asynccnt(0);
#else
  asm volatile("s_wait_asynccnt 0x0" ::: "memory");
#endif
#else
  asm volatile("s_wait_asynccnt 0x0" ::: "memory");
#endif
}

// Workgroup barrier that drains only DScnt (LDS visibility), leaving our own
// global loads in flight across the barrier (HIP __syncthreads would force
// s_wait_loadcnt 0 too, killing the cross-chunk prefetch).
__device__ __forceinline__ void wg_barrier_ds() {
  asm volatile("s_wait_dscnt 0x0\n\t"
               "s_barrier_signal -1\n\t"
               "s_barrier_wait -1" ::: "memory");
}

__device__ __forceinline__ v8f wmma16(v16h a, v16h b, v8f c) {
  // D = A(16x32 f16) * B(32x16 f16) + C(16x16 f32)
  return __builtin_amdgcn_wmma_f32_16x16x32_f16(false, a, false, b, (short)0, c,
                                                false, false);
}

// A fragment (16x32 f16): row M = lane&15, K pattern: halves[0..15] -> k0+16*hi+i.
// Two contiguous 8-half (16B) runs.
__device__ __forceinline__ v16h load_a_frag_g(const _Float16* __restrict__ base,
                                              int ld, int k0) {
  const int lane = threadIdx.x & 31;
  const int lo = lane & 15, hi = lane >> 4;
  const _Float16* r = base + (size_t)lo * ld + k0 + hi * 16;
  AFrag u;
  u.h[0] = *(const v8h*)(r);
  u.h[1] = *(const v8h*)(r + 8);
  return u.v;
}

// ---------------------------------------------------------------------------
// Weight cast + transpose: W[Kd][Nd] f32 -> Wt[Nd][Kd] f16 (tiled via LDS)
// ---------------------------------------------------------------------------
__global__ __launch_bounds__(256) void castT_kernel(const float* __restrict__ W,
                                                    _Float16* __restrict__ Wt,
                                                    int Kd, int Nd) {
  __shared__ float tile[32][33];
  const int n0 = blockIdx.x * 32, k0 = blockIdx.y * 32;
  const int tx = threadIdx.x & 31, ty = threadIdx.x >> 5;
#pragma unroll
  for (int j = 0; j < 4; ++j)
    tile[ty + j * 8][tx] = W[(size_t)(k0 + ty + j * 8) * Nd + n0 + tx];
  __syncthreads();
#pragma unroll
  for (int j = 0; j < 4; ++j)
    Wt[(size_t)(n0 + ty + j * 8) * Kd + k0 + tx] = (_Float16)tile[tx][ty + j * 8];
}

// ---------------------------------------------------------------------------
// LayerNorm over D=1024: one block (256 thr) per row, f32 in -> f16 out
// ---------------------------------------------------------------------------
__global__ __launch_bounds__(256) void ln_kernel(const float* __restrict__ x,
                                                 const float* __restrict__ w,
                                                 const float* __restrict__ b,
                                                 _Float16* __restrict__ out) {
  const int row = blockIdx.x;
  const float* xr = x + (size_t)row * Dsz;
  float v[4], s = 0.f, ss = 0.f;
#pragma unroll
  for (int j = 0; j < 4; ++j) {
    v[j] = xr[threadIdx.x + j * 256];
    s += v[j];
    ss += v[j] * v[j];
  }
#pragma unroll
  for (int off = 1; off < 32; off <<= 1) {
    s += __shfl_xor(s, off, 32);
    ss += __shfl_xor(ss, off, 32);
  }
  __shared__ float sh[16];
  const int wid = threadIdx.x >> 5, lane = threadIdx.x & 31;
  if (lane == 0) { sh[wid] = s; sh[wid + 8] = ss; }
  __syncthreads();
  float st = 0.f, sst = 0.f;
#pragma unroll
  for (int i = 0; i < 8; ++i) { st += sh[i]; sst += sh[i + 8]; }
  const float mu = st * (1.0f / Dsz);
  const float var = sst * (1.0f / Dsz) - mu * mu;
  const float rs = rsqrtf(var + 1e-5f);
#pragma unroll
  for (int j = 0; j < 4; ++j) {
    const int i = threadIdx.x + j * 256;
    out[(size_t)row * Dsz + i] = (_Float16)((v[j] - mu) * rs * w[i] + b[i]);
  }
}

// ---------------------------------------------------------------------------
// WMMA GEMM: C[M,N] = A[M,K](f16) @ Bt[N,K](f16)^T + bias, with epilogues:
//  mode 0: out f16 [M,N]
//  mode 1: out f16 transposed per head -> vt[b][h][dh][s]
//  mode 2: out f32 [M,N] = aux(f32 residual) + C + bias
//  mode 3: out f16 [M,N] = silu(aux_f16[m,n]) * (C + bias)   (SwiGLU fuse)
// Block = 256 thr (8 waves); block tile 128x64, wave tile 16x64 (4 C tiles).
// B tile (64x32) staged into double-buffered LDS (8x less L2 traffic), via
// GLOBAL_LOAD_ASYNC_TO_LDS_B128 when available, else global->reg->ds_store.
// Barrier drains DScnt (+ASYNCcnt) only, so A prefetch stays in flight.
// ---------------------------------------------------------------------------
__global__ __launch_bounds__(256) void gemm_wmma(
    const _Float16* __restrict__ A, const _Float16* __restrict__ Bt,
    const float* __restrict__ bias, const void* __restrict__ aux,
    void* __restrict__ out, int M, int N, int K, int mode) {
  __shared__ __align__(16) _Float16 Blds[2][64][40];  // 40-half padded rows

  const int wave = threadIdx.x >> 5;
  const int lane = threadIdx.x & 31;
  const int lo = lane & 15, hi = lane >> 4;
  const int m0 = blockIdx.y * 128 + wave * 16;
  const int n0 = blockIdx.x * 64;

  // staging role: thread tid stages 16B: row = tid/4 (0..63), seg = tid%4
  const int srow = threadIdx.x >> 2, sseg = threadIdx.x & 3;
  const _Float16* gsrc = Bt + (size_t)(n0 + srow) * K + sseg * 8;

  const v8f vzero = {0.f, 0.f, 0.f, 0.f, 0.f, 0.f, 0.f, 0.f};
  v8f acc[4];
#pragma unroll
  for (int t = 0; t < 4; ++t) acc[t] = vzero;

  const _Float16* Ab = A + (size_t)m0 * K;

  // prologue: stage B chunk 0, load A chunk 0
#if ASYNC_LDS
  __builtin_amdgcn_global_load_async_to_lds_b128(
      (gv4i_p)(v4i*)gsrc, (lv4i_p)(v4i*)&Blds[0][srow][sseg * 8], 0, 0);
  v16h a_cur = load_a_frag_g(Ab, K, 0);
  wait_async0();
  wg_barrier_ds();
#else
  {
    const v8h br0 = *(const v8h*)(gsrc);
    *(v8h*)(&Blds[0][srow][sseg * 8]) = br0;
  }
  v16h a_cur = load_a_frag_g(Ab, K, 0);
  wg_barrier_ds();
#endif

  const int nch = K >> 5;
  for (int c = 0; c < nch; ++c) {
    const int cur = c & 1;
    const int kpre = (c + 1 < nch) ? ((c + 1) << 5) : 0;  // uniform; dummy on last
#if ASYNC_LDS
    // async copy next B chunk straight into LDS (no VGPR roundtrip)
    __builtin_amdgcn_global_load_async_to_lds_b128(
        (gv4i_p)(v4i*)(gsrc + kpre),
        (lv4i_p)(v4i*)&Blds[cur ^ 1][srow][sseg * 8], 0, 0);
#else
    const v8h breg = *(const v8h*)(gsrc + kpre);
#endif
    const v16h a_nxt = load_a_frag_g(Ab, K, kpre);
    // compute current chunk: B fragments from LDS (k = 16*hi + i within chunk)
#pragma unroll
    for (int t = 0; t < 4; ++t) {
      AFrag bf;
      const _Float16* lr = &Blds[cur][t * 16 + lo][hi * 16];
      bf.h[0] = *(const v8h*)(lr);
      bf.h[1] = *(const v8h*)(lr + 8);
      acc[t] = wmma16(a_cur, bf.v, acc[t]);
    }
#if ASYNC_LDS
    wait_async0();
    wg_barrier_ds();
#else
    // stage next chunk into the other buffer (not read this iteration)
    *(v8h*)(&Blds[cur ^ 1][srow][sseg * 8]) = breg;
    wg_barrier_ds();
#endif
    a_cur = a_nxt;
  }

#pragma unroll
  for (int t = 0; t < 4; ++t) {
    const int n = n0 + t * 16 + lo;
    const float bv = bias ? bias[n] : 0.0f;
#pragma unroll
    for (int r = 0; r < 8; ++r) {
      const int m = m0 + r + 8 * hi;
      const float val = acc[t][r] + bv;
      const size_t idx = (size_t)m * N + n;
      if (mode == 0) {
        ((_Float16*)out)[idx] = (_Float16)val;
      } else if (mode == 1) {
        const int bb = m >> 11, s = m & (Ssz - 1);
        const int hh = n >> 6, dh = n & 63;
        ((_Float16*)out)[((size_t)((bb * Hsz + hh) * DHsz + dh)) * Ssz + s] =
            (_Float16)val;
      } else if (mode == 2) {
        ((float*)out)[idx] = ((const float*)aux)[idx] + val;
      } else {
        const float g = (float)((const _Float16*)aux)[idx];
        const float sg = g / (1.0f + __expf(-g));
        ((_Float16*)out)[idx] = (_Float16)(sg * val);
      }
    }
  }
}

// ---------------------------------------------------------------------------
// Windowed attention, flash-style. One wave per (b, h, 16-query block).
// 9 fixed key-chunks of 32 cover [s0-128, s0+143] (band is 272 wide).
// ---------------------------------------------------------------------------
__device__ __forceinline__ void score_tile(const _Float16* __restrict__ kb,
                                           const unsigned char* __restrict__ mb,
                                           v16h qa0, v16h qa1, int tcol0, int s0,
                                           float* __restrict__ p) {
  const int lane = threadIdx.x & 31;
  const int lo = lane & 15, hi = lane >> 4;
  const int t = tcol0 + lo;
  const int tcl = min(max(t, 0), Ssz - 1);
  const _Float16* kr = kb + (size_t)tcl * Dsz;
  AFrag u0, u1;
  u0.h[0] = *(const v8h*)(kr + hi * 16);
  u0.h[1] = *(const v8h*)(kr + hi * 16 + 8);
  u1.h[0] = *(const v8h*)(kr + 32 + hi * 16);
  u1.h[1] = *(const v8h*)(kr + 32 + hi * 16 + 8);
  const v8f vzero = {0.f, 0.f, 0.f, 0.f, 0.f, 0.f, 0.f, 0.f};
  v8f sacc = vzero;
  sacc = wmma16(qa0, u0.v, sacc);
  sacc = wmma16(qa1, u1.v, sacc);
  const bool tv = (t >= 0 && t < Ssz) && (mb[tcl] != 0);
#pragma unroll
  for (int r = 0; r < 8; ++r) {
    const int srow = s0 + r + 8 * hi;
    const int d = srow - t;
    const bool ok = tv && (d <= HALFW) && (-d <= HALFW);
    p[r] = ok ? sacc[r] * 0.125f : -1e9f;  // 1/sqrt(64)
  }
}

__global__ __launch_bounds__(128) void attn_wmma(
    const _Float16* __restrict__ q, const _Float16* __restrict__ k,
    const _Float16* __restrict__ vt, const unsigned char* __restrict__ mask,
    _Float16* __restrict__ ctx) {
  __shared__ _Float16 plds[4][16][32];  // per-wave P staging (C-layout -> A-layout)
  const int wid = threadIdx.x >> 5;
  const int lane = threadIdx.x & 31;
  const int lo = lane & 15, hi = lane >> 4;

  const int blk = blockIdx.x * 4 + wid;       // 0 .. B*H*(S/16)-1
  const int sb = blk & (Ssz / 16 - 1);
  const int h = (blk >> 7) & (Hsz - 1);
  const int b = blk >> 11;
  const int s0 = sb * 16;

  const _Float16* qb = q + (size_t)b * Ssz * Dsz + h * DHsz;
  const _Float16* kb = k + (size_t)b * Ssz * Dsz + h * DHsz;
  const _Float16* vb = vt + (size_t)(b * Hsz + h) * DHsz * Ssz;
  const unsigned char* mb = mask + (size_t)b * Ssz;

  const v16h qa0 = load_a_frag_g(qb + (size_t)s0 * Dsz, Dsz, 0);
  const v16h qa1 = load_a_frag_g(qb + (size_t)s0 * Dsz, Dsz, 32);

  const v8f vzero = {0.f, 0.f, 0.f, 0.f, 0.f, 0.f, 0.f, 0.f};
  v8f oacc[4];
#pragma unroll
  for (int t = 0; t < 4; ++t) oacc[t] = vzero;
  float mrow[8], lrow[8];
#pragma unroll
  for (int r = 0; r < 8; ++r) { mrow[r] = -1e30f; lrow[r] = 0.0f; }

  for (int c = 0; c < 9; ++c) {
    const int tbase = s0 - HALFW + 32 * c;

    // Preload V B-fragments early: they are consumed only after the softmax,
    // so their latency hides behind the score WMMAs + VALU softmax work.
    // Column dh = td*16+lo, K = t contiguous in vt[dh][t].
    const int tstart = tbase + 16 * hi;
    const int cl = min(max(tstart, 0), Ssz - 16);
    AFrag vf[4];
#pragma unroll
    for (int td = 0; td < 4; ++td) {
      const _Float16* vr = vb + (size_t)(td * 16 + lo) * Ssz + cl;
      vf[td].h[0] = *(const v8h*)(vr);
      vf[td].h[1] = *(const v8h*)(vr + 8);
    }

    float p0[8], p1[8];
    score_tile(kb, mb, qa0, qa1, tbase, s0, p0);
    score_tile(kb, mb, qa0, qa1, tbase + 16, s0, p1);

    float sc[8];
#pragma unroll
    for (int r = 0; r < 8; ++r) {
      float rm = fmaxf(p0[r], p1[r]);
#pragma unroll
      for (int off = 1; off < 16; off <<= 1)
        rm = fmaxf(rm, __shfl_xor(rm, off, 32));  // row lives in a 16-lane half
      const float mn = fmaxf(mrow[r], rm);
      sc[r] = __expf(mrow[r] - mn);
      mrow[r] = mn;
      // hard-zero masked entries (avoids exp(0)=1 when a whole chunk is masked)
      p0[r] = (p0[r] < -5e8f) ? 0.0f : __expf(p0[r] - mn);
      p1[r] = (p1[r] < -5e8f) ? 0.0f : __expf(p1[r] - mn);
      float rs = p0[r] + p1[r];
#pragma unroll
      for (int off = 1; off < 16; off <<= 1) rs += __shfl_xor(rs, off, 32);
      lrow[r] = lrow[r] * sc[r] + rs;
    }
#pragma unroll
    for (int t = 0; t < 4; ++t)
#pragma unroll
      for (int r = 0; r < 8; ++r) oacc[t][r] *= sc[r];

    // P (C-layout) -> LDS -> A-fragment layout
#pragma unroll
    for (int r = 0; r < 8; ++r) {
      plds[wid][r + 8 * hi][lo] = (_Float16)p0[r];
      plds[wid][r + 8 * hi][lo + 16] = (_Float16)p1[r];
    }
    AFrag pa;
    const _Float16* prow = &plds[wid][lo][0];
    pa.h[0] = *(const v8h*)(prow + 16 * hi);
    pa.h[1] = *(const v8h*)(prow + 16 * hi + 8);

#pragma unroll
    for (int td = 0; td < 4; ++td)
      oacc[td] = wmma16(pa.v, vf[td].v, oacc[td]);
  }

  _Float16* cb = ctx + (size_t)b * Ssz * Dsz + h * DHsz;
#pragma unroll
  for (int r = 0; r < 8; ++r) {
    const int srow = s0 + r + 8 * hi;
    const float inv = 1.0f / lrow[r];
#pragma unroll
    for (int td = 0; td < 4; ++td)
      cb[(size_t)srow * Dsz + td * 16 + lo] = (_Float16)(oacc[td][r] * inv);
  }
}

// ---------------------------------------------------------------------------
extern "C" void kernel_launch(void* const* d_in, const int* in_sizes, int n_in,
                              void* d_out, int out_size, void* d_ws, size_t ws_size,
                              hipStream_t stream) {
  const float* x = (const float*)d_in[0];
  const unsigned char* mask = (const unsigned char*)d_in[1];
  const float* ln1_w = (const float*)d_in[2];
  const float* ln1_b = (const float*)d_in[3];
  const float* Wq = (const float*)d_in[4];
  const float* bq = (const float*)d_in[5];
  const float* Wk = (const float*)d_in[6];
  const float* bk = (const float*)d_in[7];
  const float* Wv = (const float*)d_in[8];
  const float* bv = (const float*)d_in[9];
  const float* Wo = (const float*)d_in[10];
  const float* bo = (const float*)d_in[11];
  const float* ln2_w = (const float*)d_in[12];
  const float* ln2_b = (const float*)d_in[13];
  const float* Wg = (const float*)d_in[14];
  const float* bg = (const float*)d_in[15];
  const float* Wu = (const float*)d_in[16];
  const float* bu = (const float*)d_in[17];
  const float* Wd = (const float*)d_in[18];
  const float* bd = (const float*)d_in[19];

  char* p = (char*)d_ws;
  auto alloc = [&](size_t bytes) -> void* {
    void* r = (void*)p;
    p += (bytes + 255) & ~(size_t)255;
    return r;
  };
  const size_t DD = (size_t)Dsz * Dsz;
  const size_t DHID = (size_t)Dsz * HIDsz;
  const size_t BSD = (size_t)Bsz * Ssz * Dsz;
  const size_t BSH = (size_t)Bsz * Ssz * HIDsz;

  _Float16* WqT = (_Float16*)alloc(DD * 2);
  _Float16* WkT = (_Float16*)alloc(DD * 2);
  _Float16* WvT = (_Float16*)alloc(DD * 2);
  _Float16* WoT = (_Float16*)alloc(DD * 2);
  _Float16* WgT = (_Float16*)alloc(DHID * 2);
  _Float16* WuT = (_Float16*)alloc(DHID * 2);
  _Float16* WdT = (_Float16*)alloc(DHID * 2);
  _Float16* h1 = (_Float16*)alloc(BSD * 2);
  _Float16* qf = (_Float16*)alloc(BSD * 2);
  _Float16* kf = (_Float16*)alloc(BSD * 2);
  _Float16* vtf = (_Float16*)alloc(BSD * 2);
  _Float16* ctxf = (_Float16*)alloc(BSD * 2);
  _Float16* h2 = (_Float16*)alloc(BSD * 2);
  _Float16* gf = (_Float16*)alloc(BSH * 2);
  _Float16* actf = (_Float16*)alloc(BSH * 2);
  float* x2 = (float*)alloc(BSD * 4);

  const dim3 tb(256);
  // Weight cast + transpose to f16 [N][K]
  castT_kernel<<<dim3(Dsz / 32, Dsz / 32), tb, 0, stream>>>(Wq, WqT, Dsz, Dsz);
  castT_kernel<<<dim3(Dsz / 32, Dsz / 32), tb, 0, stream>>>(Wk, WkT, Dsz, Dsz);
  castT_kernel<<<dim3(Dsz / 32, Dsz / 32), tb, 0, stream>>>(Wv, WvT, Dsz, Dsz);
  castT_kernel<<<dim3(Dsz / 32, Dsz / 32), tb, 0, stream>>>(Wo, WoT, Dsz, Dsz);
  castT_kernel<<<dim3(HIDsz / 32, Dsz / 32), tb, 0, stream>>>(Wg, WgT, Dsz, HIDsz);
  castT_kernel<<<dim3(HIDsz / 32, Dsz / 32), tb, 0, stream>>>(Wu, WuT, Dsz, HIDsz);
  castT_kernel<<<dim3(Dsz / 32, HIDsz / 32), tb, 0, stream>>>(Wd, WdT, HIDsz, Dsz);

  const int M = Bsz * Ssz;
  const dim3 gD(Dsz / 64, M / 128);
  const dim3 gH(HIDsz / 64, M / 128);

  // LN1 -> h1 (f16)
  ln_kernel<<<M, tb, 0, stream>>>(x, ln1_w, ln1_b, h1);
  // q, k (f16 [B,S,D]); v transposed per head (f16 [b,h,dh,s])
  gemm_wmma<<<gD, tb, 0, stream>>>(h1, WqT, bq, nullptr, qf, M, Dsz, Dsz, 0);
  gemm_wmma<<<gD, tb, 0, stream>>>(h1, WkT, bk, nullptr, kf, M, Dsz, Dsz, 0);
  gemm_wmma<<<gD, tb, 0, stream>>>(h1, WvT, bv, nullptr, vtf, M, Dsz, Dsz, 1);
  // windowed flash attention -> ctx (f16)
  attn_wmma<<<(Bsz * Hsz * (Ssz / 16)) / 4, 128, 0, stream>>>(qf, kf, vtf, mask,
                                                              ctxf);
  // x2 = x + ctx @ Wo + bo  (f32)
  gemm_wmma<<<gD, tb, 0, stream>>>(ctxf, WoT, bo, x, x2, M, Dsz, Dsz, 2);
  // LN2 -> h2 (f16)
  ln_kernel<<<M, tb, 0, stream>>>(x2, ln2_w, ln2_b, h2);
  // g = h2@Wg + bg (f16); act = silu(g) * (h2@Wu + bu) (fused epilogue)
  gemm_wmma<<<gH, tb, 0, stream>>>(h2, WgT, bg, nullptr, gf, M, HIDsz, Dsz, 0);
  gemm_wmma<<<gH, tb, 0, stream>>>(h2, WuT, bu, gf, actf, M, HIDsz, Dsz, 3);
  // out = x2 + act @ Wd + bd (f32) -> d_out
  gemm_wmma<<<gD, tb, 0, stream>>>(actf, WdT, bd, x2, d_out, M, Dsz, HIDsz, 2);
}
